// HomoGAT_38319698215187
// MI455X (gfx1250) — matrix operand order repaired and checked by reading, not compiled, not run on hardware
//
#include <hip/hip_runtime.h>

#define NN 50000
#define NE 800000

typedef _Float16 h8  __attribute__((ext_vector_type(8)));
typedef _Float16 v16h __attribute__((ext_vector_type(16)));
typedef float    v8f  __attribute__((ext_vector_type(8)));

// ---------------- elementwise f32 -> f16 ----------------
__global__ void k_cvt_f16(const float* __restrict__ in, _Float16* __restrict__ out, int n) {
  int i = blockIdx.x * blockDim.x + threadIdx.x;
  if (i < n) out[i] = (_Float16)in[i];
}

// ---------------- pack W [K x 256] f32 into per-lane B-fragment order ----------------
// packed[((kt*16+nt)*32+lane)*16 + j] = W[kt*32 + (lane>=16?16:0) + j][nt*16 + (lane&15)]
__global__ void k_pack_w(const float* __restrict__ W, _Float16* __restrict__ P, int K) {
  int o = blockIdx.x * blockDim.x + threadIdx.x;
  if (o >= K * 256) return;
  int j    = o & 15;
  int lane = (o >> 4) & 31;
  int nt   = (o >> 9) & 15;
  int kt   = o >> 13;
  int kk = kt * 32 + ((lane >= 16) ? 16 : 0) + j;
  int nn = nt * 16 + (lane & 15);
  P[o] = (_Float16)W[kk * 256 + nn];
}

// ---------------- WMMA GEMM: [50000 x KDIM] f16 @ packed [KDIM x 256] -> 256 cols ----------------
// A block (16 x KDIM) is staged into LDS once per block via gfx1250 async copies
// (global_load_async_to_lds_b128 + s_wait_asynccnt), then each wave reads its
// fragments with ds_load_b128. LDS rows padded by 8 halves (16B) to rotate banks.
// MODE 0: tanh epilogue, f16 output (feeds next GEMM). MODE 1: f32 output.
template<int KDIM, int MODE>
__global__ __launch_bounds__(256) void k_gemm(const _Float16* __restrict__ A,
                                              const _Float16* __restrict__ BP,
                                              _Float16* __restrict__ Oh,
                                              float* __restrict__ Of) {
  const int lane = threadIdx.x & 31;
  const int wave = threadIdx.x >> 5;
  const int rowTile = blockIdx.x;                 // 16 rows per block, M = 3125*16 exactly
  constexpr int LROW = KDIM + 8;                  // padded LDS row stride (halves)
  __shared__ __align__(16) _Float16 shA[16 * LROW];

  // ---- async stage A block into LDS ----
  {
    const int chunksPerRow = KDIM / 8;            // b128 chunks per row
    for (int t = threadIdx.x; t < 16 * chunksPerRow; t += 256) {
      int row = t / chunksPerRow, kc = t % chunksPerRow;
      unsigned lds = (unsigned)(size_t)&shA[row * LROW + kc * 8];
      unsigned long long g =
          (unsigned long long)(A + (long)(rowTile * 16 + row) * KDIM + kc * 8);
      asm volatile("global_load_async_to_lds_b128 %0, %1, off"
                   :: "v"(lds), "v"(g) : "memory");
    }
    asm volatile("s_wait_asynccnt 0" ::: "memory");
    __syncthreads();
  }

  const int kb = (lane >= 16) ? 16 : 0;           // 16-bit A-fragment K split
  const _Float16* As = &shA[(lane & 15) * LROW];
  const int nt0 = wave * 2, nt1 = wave * 2 + 1;   // two 16-col tiles per wave
  v8f c0 = {}; v8f c1 = {};
  for (int k0 = 0; k0 < KDIM; k0 += 32) {
    h8 alo = *(const h8*)(As + k0 + kb);
    h8 ahi = *(const h8*)(As + k0 + kb + 8);
    v16h a = __builtin_shufflevector(alo, ahi, 0,1,2,3,4,5,6,7,8,9,10,11,12,13,14,15);

    const _Float16* bp0 = BP + ((((k0 >> 5) * 16 + nt0) * 32 + lane) << 4);
    h8 b0lo = *(const h8*)bp0; h8 b0hi = *(const h8*)(bp0 + 8);
    v16h b0 = __builtin_shufflevector(b0lo, b0hi, 0,1,2,3,4,5,6,7,8,9,10,11,12,13,14,15);
    c0 = __builtin_amdgcn_wmma_f32_16x16x32_f16(false, a, false, b0, (short)0, c0, false, false);

    const _Float16* bp1 = BP + ((((k0 >> 5) * 16 + nt1) * 32 + lane) << 4);
    h8 b1lo = *(const h8*)bp1; h8 b1hi = *(const h8*)(bp1 + 8);
    v16h b1 = __builtin_shufflevector(b1lo, b1hi, 0,1,2,3,4,5,6,7,8,9,10,11,12,13,14,15);
    c1 = __builtin_amdgcn_wmma_f32_16x16x32_f16(false, a, false, b1, (short)0, c1, false, false);
  }
  // C/D layout: VGPR v -> M = v (lanes 0-15) or v+8 (lanes 16-31); N = lane&15
  const int rbase = rowTile * 16 + ((lane >= 16) ? 8 : 0);
  const int col0 = nt0 * 16 + (lane & 15);
  const int col1 = nt1 * 16 + (lane & 15);
#pragma unroll
  for (int v = 0; v < 8; ++v) {
    int rr = rbase + v;
    if (MODE == 0) {
      Oh[(long)rr * 256 + col0] = (_Float16)tanhf(c0[v]);
      Oh[(long)rr * 256 + col1] = (_Float16)tanhf(c1[v]);
    } else {
      Of[(long)rr * 256 + col0] = c0[v];
      Of[(long)rr * 256 + col1] = c1[v];
    }
  }
}

// ---------------- per-node attention scores el/er (one wave per node) ----------------
__global__ void k_attn(const float* __restrict__ feat, const float* __restrict__ al,
                       const float* __restrict__ ar, float* __restrict__ el,
                       float* __restrict__ er) {
  int lane = threadIdx.x & 31;
  int node = blockIdx.x * (blockDim.x >> 5) + (threadIdx.x >> 5);
  if (node >= NN) return;
  const float* f = feat + (long)node * 256;
  float pl[4] = {0, 0, 0, 0}, pr[4] = {0, 0, 0, 0};
#pragma unroll
  for (int k = 0; k < 8; ++k) {
    int i = k * 32 + lane;           // head = i>>6 = k>>1
    float v = f[i];
    pl[k >> 1] += v * al[i];
    pr[k >> 1] += v * ar[i];
  }
#pragma unroll
  for (int h = 0; h < 4; ++h) {
    float a = pl[h], b = pr[h];
    for (int off = 16; off; off >>= 1) { a += __shfl_xor(a, off, 32); b += __shfl_xor(b, off, 32); }
    if (lane == 0) { el[node * 4 + h] = a; er[node * 4 + h] = b; }
  }
}

// ---------------- CSR-by-dst build ----------------
__global__ void k_hist(const int* __restrict__ dst, int* __restrict__ rp) {
  int e = blockIdx.x * blockDim.x + threadIdx.x;
  if (e < NE) atomicAdd(&rp[dst[e] + 1], 1);
}

__global__ void k_scan(int* __restrict__ a, int n) {   // single-block inclusive scan
  __shared__ int tmp[1024];
  __shared__ int carry;
  if (threadIdx.x == 0) carry = 0;
  __syncthreads();
  for (int base = 0; base < n; base += 1024) {
    int i = base + threadIdx.x;
    int v = (i < n) ? a[i] : 0;
    tmp[threadIdx.x] = v;
    __syncthreads();
    for (int off = 1; off < 1024; off <<= 1) {
      int t = (threadIdx.x >= (unsigned)off) ? tmp[threadIdx.x - off] : 0;
      __syncthreads();
      tmp[threadIdx.x] += t;
      __syncthreads();
    }
    int c = carry;
    if (i < n) a[i] = tmp[threadIdx.x] + c;
    __syncthreads();
    if (threadIdx.x == 1023) carry = c + tmp[1023];
    __syncthreads();
  }
}

__global__ void k_fill(const int* __restrict__ dst, const int* __restrict__ rp,
                       int* __restrict__ cnt, int* __restrict__ csr) {
  int e = blockIdx.x * blockDim.x + threadIdx.x;
  if (e < NE) {
    int d = dst[e];
    int p = atomicAdd(&cnt[d], 1);
    csr[rp[d] + p] = e;
  }
}

// ---------------- per-node online-softmax aggregation (one wave per node) ----------------
// MODE 0: layer1 -> ELU(out+bias) stored f16 (feeds next GEMM)
// MODE 1: layer2 -> ELU(out+bias), mean over 4 heads -> final f32 [N,64]
template<int MODE>
__global__ void k_agg(const float* __restrict__ feat, const float* __restrict__ el,
                      const float* __restrict__ er, const int* __restrict__ src,
                      const int* __restrict__ rp, const int* __restrict__ csr,
                      const float* __restrict__ bias,
                      _Float16* __restrict__ outh, float* __restrict__ outf) {
  int lane = threadIdx.x & 31;
  int node = blockIdx.x * (blockDim.x >> 5) + (threadIdx.x >> 5);
  if (node >= NN) return;
  float m[4] = {-1e30f, -1e30f, -1e30f, -1e30f};
  float z[4] = {0, 0, 0, 0};
  float acc[8] = {0, 0, 0, 0, 0, 0, 0, 0};
  float ern[4];
#pragma unroll
  for (int h = 0; h < 4; ++h) ern[h] = er[node * 4 + h];
  int beg = rp[node], end = rp[node + 1];
  for (int j = beg; j < end; ++j) {
    int e = csr[j];
    int s = src[e];
    const float* fs = feat + (long)s * 256;
    if (j + 1 < end) {                 // prefetch next edge's feature row (gfx1250 global_prefetch)
      int s2 = src[csr[j + 1]];
      __builtin_prefetch(feat + (long)s2 * 256 + lane * 8, 0, 1);
    }
    float scale[4], w[4];
#pragma unroll
    for (int h = 0; h < 4; ++h) {
      float sc = el[s * 4 + h] + ern[h];
      sc = sc > 0.f ? sc : 0.2f * sc;          // leaky_relu(0.2)
      float mn = fmaxf(m[h], sc);
      scale[h] = __expf(m[h] - mn);
      w[h] = __expf(sc - mn);
      z[h] = z[h] * scale[h] + w[h];
      m[h] = mn;
    }
#pragma unroll
    for (int k = 0; k < 8; ++k) {
      float f = fs[k * 32 + lane];
      int h = k >> 1;
      acc[k] = acc[k] * scale[h] + w[h] * f;
    }
  }
  float v[8];
#pragma unroll
  for (int k = 0; k < 8; ++k) {
    int h = k >> 1;
    float t = (z[h] > 0.f) ? acc[k] / z[h] : 0.f;
    t += bias[k * 32 + lane];
    v[k] = t > 0.f ? t : (__expf(t) - 1.f);    // ELU
  }
  if (MODE == 0) {
#pragma unroll
    for (int k = 0; k < 8; ++k) outh[(long)node * 256 + k * 32 + lane] = (_Float16)v[k];
  } else {
    outf[(long)node * 64 + lane]      = 0.25f * (v[0] + v[2] + v[4] + v[6]);
    outf[(long)node * 64 + 32 + lane] = 0.25f * (v[1] + v[3] + v[5] + v[7]);
  }
}

// ---------------- host side ----------------
extern "C" void kernel_launch(void* const* d_in, const int* in_sizes, int n_in,
                              void* d_out, int out_size, void* d_ws, size_t ws_size,
                              hipStream_t stream) {
  const float* x   = (const float*)d_in[0];
  const float* Wfc = (const float*)d_in[1];
  const float* W1  = (const float*)d_in[2];
  const float* al1 = (const float*)d_in[3];
  const float* ar1 = (const float*)d_in[4];
  const float* b1  = (const float*)d_in[5];
  const float* W2  = (const float*)d_in[6];
  const float* al2 = (const float*)d_in[7];
  const float* ar2 = (const float*)d_in[8];
  const float* b2  = (const float*)d_in[9];
  const int*   src = (const int*)d_in[10];
  const int*   dst = (const int*)d_in[11];
  float* out = (float*)d_out;

  char* p = (char*)d_ws;
  auto carve = [&](size_t bytes) { char* r = p; p += (bytes + 255) & ~(size_t)255; return r; };
  _Float16* xh   = (_Float16*)carve((size_t)NN * 128 * 2);
  _Float16* hbuf = (_Float16*)carve((size_t)NN * 256 * 2);
  _Float16* g1h  = (_Float16*)carve((size_t)NN * 256 * 2);
  float*    feat = (float*)   carve((size_t)NN * 256 * 4);
  float*    el   = (float*)   carve((size_t)NN * 4 * 4);
  float*    er   = (float*)   carve((size_t)NN * 4 * 4);
  _Float16* WfcP = (_Float16*)carve((size_t)128 * 256 * 2);
  _Float16* W1P  = (_Float16*)carve((size_t)256 * 256 * 2);
  _Float16* W2P  = (_Float16*)carve((size_t)256 * 256 * 2);
  int* rpn  = (int*)carve((size_t)(NN + 1) * 4);
  int* cnt  = (int*)carve((size_t)NN * 4);
  int* csr  = (int*)carve((size_t)NE * 4);

  // ---- CSR by dst (shared by both layers) ----
  hipMemsetAsync(rpn, 0, (NN + 1) * 4, stream);
  hipMemsetAsync(cnt, 0, NN * 4, stream);
  k_hist<<<(NE + 255) / 256, 256, 0, stream>>>(dst, rpn);
  k_scan<<<1, 1024, 0, stream>>>(rpn, NN + 1);
  k_fill<<<(NE + 255) / 256, 256, 0, stream>>>(dst, rpn, cnt, csr);

  // ---- precision conversion + weight packing ----
  k_cvt_f16<<<(NN * 128 + 255) / 256, 256, 0, stream>>>(x, xh, NN * 128);
  k_pack_w<<<(128 * 256 + 255) / 256, 256, 0, stream>>>(Wfc, WfcP, 128);
  k_pack_w<<<(256 * 256 + 255) / 256, 256, 0, stream>>>(W1, W1P, 256);
  k_pack_w<<<(256 * 256 + 255) / 256, 256, 0, stream>>>(W2, W2P, 256);

  // ---- h = tanh(x @ Wfc) ----
  k_gemm<128, 0><<<NN / 16, 256, 0, stream>>>(xh, WfcP, hbuf, nullptr);

  // ---- layer 1 ----
  k_gemm<256, 1><<<NN / 16, 256, 0, stream>>>(hbuf, W1P, nullptr, feat);
  k_attn<<<NN / 8, 256, 0, stream>>>(feat, al1, ar1, el, er);
  k_agg<0><<<NN / 8, 256, 0, stream>>>(feat, el, er, src, rpn, csr, b1, g1h, nullptr);

  // ---- layer 2 (+ fused head-mean to d_out) ----
  k_gemm<256, 1><<<NN / 16, 256, 0, stream>>>(g1h, W2P, nullptr, feat);
  k_attn<<<NN / 8, 256, 0, stream>>>(feat, al2, ar2, el, er);
  k_agg<1><<<NN / 8, 256, 0, stream>>>(feat, el, er, src, rpn, csr, b2, nullptr, out);
}
